// SpikingMLP_79276506349653
// MI455X (gfx1250) — matrix operand order repaired and checked by reading, not compiled, not run on hardware
//
#include <hip/hip_runtime.h>
#include <hip/hip_bf16.h>
#include <stdint.h>

typedef __bf16 bf16;
typedef __attribute__((ext_vector_type(16))) __bf16 v16bf;
typedef __attribute__((ext_vector_type(8)))  __bf16 v8bf;
typedef __attribute__((ext_vector_type(8)))  float  v8f;
typedef __attribute__((ext_vector_type(4)))  unsigned int v4u;
typedef __attribute__((ext_vector_type(8)))  int v8i;
typedef __attribute__((ext_vector_type(4)))  int v4i;

union Frag16 { v16bf v; v8bf h[2]; };

// ---------------------------------------------------------------------------
// fp32 -> bf16 conversion (grid-stride)
// ---------------------------------------------------------------------------
__global__ __launch_bounds__(256) void f32_to_bf16_kernel(
    const float* __restrict__ in, bf16* __restrict__ out, int n) {
  int i = blockIdx.x * blockDim.x + threadIdx.x;
  int stride = gridDim.x * blockDim.x;
  for (; i < n; i += stride) out[i] = (bf16)in[i];
}

// ---------------------------------------------------------------------------
// TDM 2D tile load: rows x 64 bf16 elements (128B rows), LDS pitch = 72 bf16
// (144B) via TDM pad feature (pad_interval code 4 = 32 DWORDs per row, then
// pad_amount code 3 = 4 DWORDs = 16B).
// ---------------------------------------------------------------------------
__device__ inline void tdm_load_tile64(uint32_t lds_byte_off, const bf16* gptr,
                                       uint32_t rows, uint32_t src_stride_elems) {
  uint64_t ga = (uint64_t)(uintptr_t)gptr;
  v4u g0;
  g0.x = 1u;                                   // count=1, user descriptor
  g0.y = lds_byte_off;                         // lds_addr (bytes)
  g0.z = (uint32_t)ga;                         // global_addr[31:0]
  g0.w = (uint32_t)((ga >> 32) & 0x01FFFFFFu) | (2u << 30);  // addr[56:32] | type=2
  const uint32_t dim0 = 64u;                   // tensor_dim0 == tile_dim0 (elems)
  const uint32_t d0 = (1u << 16)   /* data_size = 2 bytes */
                    | (1u << 20)   /* pad_enable */
                    | (4u << 22)   /* pad_interval: 32 DWORDs (=128B row) */
                    | (3u << 25);  /* pad_amount: 4 DWORDs (=16B) */
  v8i g1 = {
    (int)d0,
    (int)(dim0 << 16),                                  // tensor_dim0[15:0]
    (int)((dim0 >> 16) | ((rows & 0xFFFFu) << 16)),     // dim0[31:16] | tensor_dim1[15:0]
    (int)(((rows >> 16) & 0xFFFFu) | (dim0 << 16)),     // dim1[31:16] | tile_dim0=64
    (int)(rows & 0xFFFFu),                              // tile_dim1 | tile_dim2=0
    (int)src_stride_elems,                              // tensor_dim0_stride[31:0]
    0, 0
  };
  v4i gz = {0, 0, 0, 0};
#if defined(__clang_major__) && (__clang_major__ >= 23)
  v8i gz8 = {0, 0, 0, 0, 0, 0, 0, 0};
  __builtin_amdgcn_tensor_load_to_lds(g0, g1, gz, gz, gz8, 0);
#else
  __builtin_amdgcn_tensor_load_to_lds(g0, g1, gz, gz, 0);
#endif
}

// ---------------------------------------------------------------------------
// GEMM: C[M,N] (f32) = A[M,K] (bf16, row-major) * B[N,K]^T (bf16, row-major)
// Block tile 64x256, 8 waves (2x4), wave tile 32x64, K-chunk 64 (2 WMMA
// k-steps per chunk). Double-buffered TDM pipeline: issue chunk k+1, then
// s_wait_tensorcnt(2) guarantees chunk k's two DMAs landed (in-order counter).
// ---------------------------------------------------------------------------
#define BM 64
#define BN 256
#define BK 64
#define LDSP 72            // LDS pitch in bf16 elements (64 + 8 pad)
#define ASZ (BM * LDSP)    // 4608 elems / 9216 B per buffer
#define BSZ (BN * LDSP)    // 18432 elems / 36864 B per buffer

struct WaveCtx { int hi16, idx, wm, wn; };

__device__ inline void wmma_chunk(const bf16* __restrict__ curA,
                                  const bf16* __restrict__ curB,
                                  const WaveCtx& w, v8f acc[2][4]) {
#pragma unroll
  for (int s = 0; s < 2; ++s) {  // two K=32 sub-steps of the K=64 chunk
    const int ko = s * 32;
    Frag16 af[2], bfr[4];
#pragma unroll
    for (int i = 0; i < 2; ++i) {
      // A 16-bit 16x32 layout: lanes 0-15 hold K{0..7,16..23}, lanes 16-31 K{8..15,24..31}
      const bf16* p = &curA[(w.wm * 32 + i * 16 + w.idx) * LDSP + ko];
      af[i].h[0] = *(const v8bf*)(p + w.hi16 * 8);
      af[i].h[1] = *(const v8bf*)(p + 16 + w.hi16 * 8);
    }
#pragma unroll
    for (int j = 0; j < 4; ++j) {
      // B 16-bit 32x16 layout: lanes 0-15 hold K{0..7,16..23}, lanes 16-31 K{8..15,24..31}
      const bf16* p = &curB[(w.wn * 64 + j * 16 + w.idx) * LDSP + ko];
      bfr[j].h[0] = *(const v8bf*)(p + w.hi16 * 8);
      bfr[j].h[1] = *(const v8bf*)(p + 16 + w.hi16 * 8);
    }
#pragma unroll
    for (int i = 0; i < 2; ++i)
#pragma unroll
      for (int j = 0; j < 4; ++j)
        acc[i][j] = __builtin_amdgcn_wmma_f32_16x16x32_bf16(
            false, af[i].v, false, bfr[j].v, (short)0, acc[i][j], false, false);
  }
}

__global__ __launch_bounds__(256) void gemm_bf16_wmma_kernel(
    const bf16* __restrict__ A, const bf16* __restrict__ Bw,
    float* __restrict__ C, int M, int N, int K) {
  __shared__ bf16 As[2 * ASZ];
  __shared__ bf16 Bs[2 * BSZ];

  const int tid  = threadIdx.x;
  const int lane = tid & 31;
  const int wid  = tid >> 5;
  WaveCtx w;
  w.hi16 = lane >> 4;
  w.idx  = lane & 15;
  w.wm   = wid >> 2;   // 0..1
  w.wn   = wid & 3;    // 0..3
  const int m0 = blockIdx.y * BM;
  const int n0 = blockIdx.x * BN;

  const bf16* Abase = A + (size_t)m0 * K;
  const bf16* Bbase = Bw + (size_t)n0 * K;

  v8f acc[2][4] = {};

  const int nk = K / BK;
  // Prologue: fill buffer 0
  if (tid < 32) {
    tdm_load_tile64((uint32_t)(uintptr_t)&As[0], Abase, BM, (uint32_t)K);
    tdm_load_tile64((uint32_t)(uintptr_t)&Bs[0], Bbase, BN, (uint32_t)K);
  }
  for (int kc = 0; kc < nk - 1; ++kc) {
    const int cur = kc & 1;
    if (tid < 32) {  // wave 0 prefetches next chunk into the other buffer
      const int k1 = (kc + 1) * BK;
      tdm_load_tile64((uint32_t)(uintptr_t)&As[(cur ^ 1) * ASZ], Abase + k1,
                      BM, (uint32_t)K);
      tdm_load_tile64((uint32_t)(uintptr_t)&Bs[(cur ^ 1) * BSZ], Bbase + k1,
                      BN, (uint32_t)K);
    }
    __builtin_amdgcn_s_wait_tensorcnt(2);  // current buffer's 2 DMAs complete
    __syncthreads();
    wmma_chunk(&As[cur * ASZ], &Bs[cur * BSZ], w, acc);
    __syncthreads();  // done reading; next iter may overwrite this buffer
  }
  __builtin_amdgcn_s_wait_tensorcnt(0);
  __syncthreads();
  wmma_chunk(&As[((nk - 1) & 1) * ASZ], &Bs[((nk - 1) & 1) * BSZ], w, acc);

  // Epilogue: C/D layout — VGPR r: lanes0-15 M=r, lanes16-31 M=8+r, N=lane&15
#pragma unroll
  for (int i = 0; i < 2; ++i)
#pragma unroll
    for (int j = 0; j < 4; ++j) {
      const int mbase = m0 + w.wm * 32 + i * 16 + w.hi16 * 8;
      const int nn = n0 + w.wn * 64 + j * 16 + w.idx;
      float* cp = C + (size_t)mbase * N + nn;
#pragma unroll
      for (int r = 0; r < 8; ++r) cp[(size_t)r * N] = acc[i][j][r];
    }
}

// ---------------------------------------------------------------------------
// Fused bias + LayerNorm + LIF over T. One block per (b,l) pair; membrane
// potential lives in registers across the serial T loop. C = NCOL*256.
// ---------------------------------------------------------------------------
template <int NCOL, typename OutT>
__global__ __launch_bounds__(256) void ln_lif_kernel(
    const float* __restrict__ X, const float* __restrict__ bias,
    const float* __restrict__ g, const float* __restrict__ be,
    OutT* __restrict__ out, int T, int L, int C) {
  const int tid = threadIdx.x;
  const int b = blockIdx.x / L;
  const int l = blockIdx.x % L;
  __shared__ float red[16];

  float vmem[NCOL], gg[NCOL], bb[NCOL], bs[NCOL];
#pragma unroll
  for (int c = 0; c < NCOL; ++c) {
    int col = c * 256 + tid;
    vmem[c] = 0.0f; gg[c] = g[col]; bb[c] = be[col]; bs[c] = bias[col];
  }
  const float invC = 1.0f / (float)C;

  for (int t = 0; t < T; ++t) {
    const size_t row = ((size_t)b * T + t) * L + l;
    const float* p = X + row * C;
    float vals[NCOL], s = 0.0f, q = 0.0f;
#pragma unroll
    for (int c = 0; c < NCOL; ++c) {
      float v = p[c * 256 + tid] + bs[c];
      vals[c] = v; s += v; q += v * v;
    }
#pragma unroll
    for (int off = 16; off >= 1; off >>= 1) {
      s += __shfl_xor(s, off);
      q += __shfl_xor(q, off);
    }
    if ((tid & 31) == 0) { red[tid >> 5] = s; red[8 + (tid >> 5)] = q; }
    __syncthreads();
    s = 0.0f; q = 0.0f;
#pragma unroll
    for (int wv = 0; wv < 8; ++wv) { s += red[wv]; q += red[8 + wv]; }
    __syncthreads();
    const float mu = s * invC;
    const float var = q * invC - mu * mu;
    const float rstd = rsqrtf(var + 1e-5f);
    OutT* po = out + row * C;
#pragma unroll
    for (int c = 0; c < NCOL; ++c) {
      float y = (vals[c] - mu) * rstd * gg[c] + bb[c];
      float v = vmem[c] + (y - vmem[c]) * 0.5f;   // 1/TAU = 0.5
      float sp = (v >= 1.0f) ? 1.0f : 0.0f;       // heaviside(v - V_TH)
      vmem[c] = v * (1.0f - sp);                  // hard reset
      po[c * 256 + tid] = (OutT)sp;               // spikes {0,1} exact in bf16
    }
  }
}

// ---------------------------------------------------------------------------
extern "C" void kernel_launch(void* const* d_in, const int* in_sizes, int n_in,
                              void* d_out, int out_size, void* d_ws, size_t ws_size,
                              hipStream_t stream) {
  constexpr int B = 16, T = 4, L = 256, D = 768, H = 3072;
  constexpr int ROWS = B * T * L;  // 16384, row order (b,t,l) == x memory order

  const float* x   = (const float*)d_in[0];
  const float* W1  = (const float*)d_in[1];
  const float* b1  = (const float*)d_in[2];
  const float* g1  = (const float*)d_in[3];
  const float* be1 = (const float*)d_in[4];
  const float* W2  = (const float*)d_in[5];
  const float* b2  = (const float*)d_in[6];
  const float* g2  = (const float*)d_in[7];
  const float* be2 = (const float*)d_in[8];

  char* ws = (char*)d_ws;
  auto align256 = [](size_t v) { return (v + 255) & ~(size_t)255; };
  size_t off = 0;
  bf16* w1b = (bf16*)(ws + off); off += align256((size_t)H * D * sizeof(bf16));
  bf16* w2b = (bf16*)(ws + off); off += align256((size_t)D * H * sizeof(bf16));
  bf16* xb  = (bf16*)(ws + off); off += align256((size_t)ROWS * D * sizeof(bf16));
  float* h  = (float*)(ws + off); off += align256((size_t)ROWS * H * sizeof(float));
  bf16* s1b = (bf16*)(ws + off);
  float* o  = h;  // h is dead after LN1+LIF1; reuse for GEMM2 output

  f32_to_bf16_kernel<<<2048, 256, 0, stream>>>(W1, w1b, H * D);
  f32_to_bf16_kernel<<<2048, 256, 0, stream>>>(W2, w2b, D * H);
  f32_to_bf16_kernel<<<4096, 256, 0, stream>>>(x, xb, ROWS * D);

  // h = x @ W1^T   (M=16384, N=3072, K=768)
  gemm_bf16_wmma_kernel<<<dim3(H / BN, ROWS / BM), 256, 0, stream>>>(
      xb, w1b, h, ROWS, H, D);

  // s1 = LIF(LN(h + b1))  -> bf16 spikes
  ln_lif_kernel<12, bf16><<<B * L, 256, 0, stream>>>(h, b1, g1, be1, s1b, T, L, H);

  // o = s1 @ W2^T  (M=16384, N=768, K=3072)
  gemm_bf16_wmma_kernel<<<dim3(D / BN, ROWS / BM), 256, 0, stream>>>(
      s1b, w2b, o, ROWS, D, H);

  // out = LIF(LN(o + b2)) -> fp32 spikes, already in (B,T,L,D) order
  ln_lif_kernel<3, float><<<B * L, 256, 0, stream>>>(o, b2, g2, be2,
                                                     (float*)d_out, T, L, D);
}
